// AtomCrossAttDecoder_84731114816222
// MI455X (gfx1250) — matrix operand order repaired and checked by reading, not compiled, not run on hardware
//
#include <hip/hip_runtime.h>
#include <hip/hip_bf16.h>

// ---------------- dims (match reference) ----------------
#define TT 1024   // tokens
#define AA 24     // atoms per token
#define SS 512    // subsets
#define QQ 32     // queries per subset
#define KK 128    // keys per subset
#define CC 128    // atom channels
#define CTT 768   // token channels
#define CPP 16    // pair channels
#define NBB 3     // blocks
#define NHH 4     // heads
#define HDD 32    // head dim
#define FHH 256   // transition hidden (2*C)

typedef __attribute__((ext_vector_type(16))) _Float16 v16h;
typedef __attribute__((ext_vector_type(8)))  _Float16 v8h;
typedef __attribute__((ext_vector_type(4)))  _Float16 v4h;
typedef __attribute__((ext_vector_type(8)))  float    v8f;

#define WMMA_F16(a, b, c) \
  __builtin_amdgcn_wmma_f32_16x16x32_f16(false, (a), false, (b), (short)0, (c), false, false)

// ---------------- WMMA fragment helpers (CDNA5 16x16x32 f16 layouts) -------
// A (16x32 f16), row-major LDS tile (row stride 16B-aligned):
// lane L<16 -> row m0+L, runs K={k0..k0+7, k0+16..k0+23}; lanes>=16 shift +8.
// Both runs contiguous -> two ds_load_b128.
__device__ __forceinline__ v16h load_frag_a(const _Float16* As, int lda, int m0, int k0, int lane) {
  const _Float16* p = As + (m0 + (lane & 15)) * lda + k0 + ((lane >> 4) << 3);
  v8h lo = *(const v8h*)p;
  v8h hi = *(const v8h*)(p + 16);
  return __builtin_shufflevector(lo, hi, 0, 1, 2, 3, 4, 5, 6, 7, 8, 9, 10, 11, 12, 13, 14, 15);
}
// B (32x16 f16) from an N-major (transposed) LDS tile BsT[n*ldb + k]:
// lane L<16 -> col n0+L, K={k0..k0+7, k0+16..k0+23}; lanes>=16 shift +8.
__device__ __forceinline__ v16h load_frag_b(const _Float16* BsT, int ldb, int k0, int n0, int lane) {
  const _Float16* p = BsT + (n0 + (lane & 15)) * ldb + k0 + ((lane >> 4) << 3);
  v8h lo = *(const v8h*)p;
  v8h hi = *(const v8h*)(p + 16);
  return __builtin_shufflevector(lo, hi, 0, 1, 2, 3, 4, 5, 6, 7, 8, 9, 10, 11, 12, 13, 14, 15);
}
// C/D (16x16 f32, 8 VGPRs): VGPR r -> M = r (lanes 0-15) / 8+r (lanes 16-31), N = lane&15.
__device__ __forceinline__ void store_frag_lds(float* Ds, int ldd, int m0, int n0, v8f d, int lane) {
  int n = n0 + (lane & 15);
  int m = m0 + ((lane >> 4) << 3);
#pragma unroll
  for (int r = 0; r < 8; ++r) Ds[(m + r) * ldd + n] = d[r];
}

__device__ __forceinline__ v4h cvt4(float a, float b, float c, float d) {
  v4h h = { (_Float16)a, (_Float16)b, (_Float16)c, (_Float16)d };
  return h;
}

// ---------------- generic WMMA GEMM: D[M,N] = A[M,K] @ B[K,N] (fp32 io) ----
// BM=BN=64, BK=32; 128 threads = 4 waves in 2x2; each wave owns a 32x32 tile.
// Double-buffered LDS with register staging: global loads for tile k+1 issue
// before the WMMAs of tile k, converts/stores land after them.
__global__ __launch_bounds__(128) void wmma_gemm_kernel(
    const float* __restrict__ A, const float* __restrict__ B, float* __restrict__ D,
    int M, int N, int Kd) {
  __shared__ _Float16 As[2][64 * 40];    // row-major 64x32, lda=40 (80B rows)
  __shared__ _Float16 BsT[2][64 * 40];   // N-major 64x32, ldb=40
  const int tid = threadIdx.x, lane = tid & 31, wave = tid >> 5;
  const int wr = (wave >> 1) * 32, wc = (wave & 1) * 32;
  const long mBase = (long)blockIdx.y * 64, nBase = (long)blockIdx.x * 64;

  // per-thread staging coordinates (4 groups each for A and B)
  int rA[4], cA[4], nB[4], kB[4];
#pragma unroll
  for (int i = 0; i < 4; ++i) {
    int t = tid + i * 128;
    rA[i] = t >> 3; cA[i] = (t & 7) << 2;     // A: row, col4 (float4 along K)
    nB[i] = t & 63; kB[i] = (t >> 6) << 2;    // B: col, k4 (4 strided rows)
  }
  float4 fa[4];
  float  fb[4][4];
  auto load_tile = [&](int kb) {
#pragma unroll
    for (int i = 0; i < 4; ++i)
      fa[i] = *(const float4*)&A[(mBase + rA[i]) * (long)Kd + kb + cA[i]];
#pragma unroll
    for (int i = 0; i < 4; ++i) {
      const float* bp = &B[(long)(kb + kB[i]) * N + nBase + nB[i]];
#pragma unroll
      for (int j = 0; j < 4; ++j) fb[i][j] = bp[(long)j * N];
    }
  };
  auto store_tile = [&](int buf) {
#pragma unroll
    for (int i = 0; i < 4; ++i)
      *(v4h*)&As[buf][rA[i] * 40 + cA[i]] = cvt4(fa[i].x, fa[i].y, fa[i].z, fa[i].w);
#pragma unroll
    for (int i = 0; i < 4; ++i)
      *(v4h*)&BsT[buf][nB[i] * 40 + kB[i]] = cvt4(fb[i][0], fb[i][1], fb[i][2], fb[i][3]);
  };

  v8f acc[2][2] = {};
  load_tile(0);
  store_tile(0);
  int cur = 0;
  for (int kb = 0; kb < Kd; kb += 32) {
    __syncthreads();
    const bool more = (kb + 32) < Kd;
    if (more) load_tile(kb + 32);   // issue global loads early (latency hidden by WMMAs)
    v16h a0 = load_frag_a(As[cur], 40, wr + 0, 0, lane);
    v16h a1 = load_frag_a(As[cur], 40, wr + 16, 0, lane);
    v16h b0 = load_frag_b(BsT[cur], 40, 0, wc + 0, lane);
    v16h b1 = load_frag_b(BsT[cur], 40, 0, wc + 16, lane);
    acc[0][0] = WMMA_F16(a0, b0, acc[0][0]);
    acc[0][1] = WMMA_F16(a0, b1, acc[0][1]);
    acc[1][0] = WMMA_F16(a1, b0, acc[1][0]);
    acc[1][1] = WMMA_F16(a1, b1, acc[1][1]);
    if (more) store_tile(cur ^ 1);  // convert + store into the idle buffer
    cur ^= 1;
  }
  const int n = lane & 15, mb = (lane >> 4) << 3;
#pragma unroll
  for (int i = 0; i < 2; ++i)
#pragma unroll
    for (int j = 0; j < 2; ++j) {
      long m0 = mBase + wr + i * 16 + mb;
      long n0 = nBase + wc + j * 16 + n;
#pragma unroll
      for (int r = 0; r < 8; ++r) D[(m0 + r) * N + n0] = acc[i][j][r];
    }
}

// ---------------- fused attention per (subset, head) ----------------------
// logits = (Q Kt)/sqrt(HD) + pair_bias, mask, softmax, O = P V. One wave/WG.
__global__ __launch_bounds__(32) void attn_kernel(
    const float* __restrict__ qg, const float* __restrict__ kg, const float* __restrict__ vg,
    const float* __restrict__ pl, const unsigned char* __restrict__ keys_mask,
    float* __restrict__ og, int bIdx) {
  __shared__ _Float16 qf[32 * 40];    // Q row-major 32x32 (A matrix)
  __shared__ _Float16 kT[128 * 40];   // logits B (k-dim=HD, n=key): n-major kT[kk*40 + d]
  __shared__ _Float16 vT[32 * 136];   // PV B (k-dim=key, n=HD): n-major vT[d*136 + kk]
  __shared__ float    lg[32 * 128];
  __shared__ _Float16 pf[32 * 136];   // probs row-major 32x128 (A matrix for PV)
  const int s = blockIdx.x >> 2, h = blockIdx.x & 3;
  const int lane = threadIdx.x;
  const float* qs = qg + (long)s * QQ * CC + h * HDD;
  const float* ks = kg + (long)s * KK * CC + h * HDD;
  const float* vs = vg + (long)s * KK * CC + h * HDD;
  // stage Q (8 float4 groups/lane): batch loads, then converts/stores
  {
    float4 f[8];
#pragma unroll
    for (int i = 0; i < 8; ++i) {
      int t = lane + i * 32, m = t >> 3, d4 = (t & 7) << 2;
      f[i] = *(const float4*)&qs[m * CC + d4];
    }
#pragma unroll
    for (int i = 0; i < 8; ++i) {
      int t = lane + i * 32, m = t >> 3, d4 = (t & 7) << 2;
      *(v4h*)&qf[m * 40 + d4] = cvt4(f[i].x, f[i].y, f[i].z, f[i].w);
    }
  }
  // stage K (contiguous float4 along d -> kT[kk*40 + d4]); chunks of 8 groups
  for (int ch = 0; ch < 4; ++ch) {
    float4 f[8];
#pragma unroll
    for (int i = 0; i < 8; ++i) {
      int t = lane + (ch * 8 + i) * 32, kk = t >> 3, d4 = (t & 7) << 2;
      f[i] = *(const float4*)&ks[kk * CC + d4];
    }
#pragma unroll
    for (int i = 0; i < 8; ++i) {
      int t = lane + (ch * 8 + i) * 32, kk = t >> 3, d4 = (t & 7) << 2;
      *(v4h*)&kT[kk * 40 + d4] = cvt4(f[i].x, f[i].y, f[i].z, f[i].w);
    }
  }
  // stage V transposed (4 strided b32 along kk, coalesced over d) -> vT[d*136 + k4]
  for (int ch = 0; ch < 4; ++ch) {
    float f[8][4];
#pragma unroll
    for (int i = 0; i < 8; ++i) {
      int t = lane + (ch * 8 + i) * 32, d = t & 31, k4 = (t >> 5) << 2;
      const float* vp = &vs[k4 * CC + d];
#pragma unroll
      for (int j = 0; j < 4; ++j) f[i][j] = vp[j * CC];
    }
#pragma unroll
    for (int i = 0; i < 8; ++i) {
      int t = lane + (ch * 8 + i) * 32, d = t & 31, k4 = (t >> 5) << 2;
      *(v4h*)&vT[d * 136 + k4] = cvt4(f[i][0], f[i][1], f[i][2], f[i][3]);
    }
  }
  __syncthreads();
  // logits: M=32 (2 tiles) x N=128 (8 tiles), single K-step of 32 (=HD).
  {
    v16h a0 = load_frag_a(qf, 40, 0, 0, lane);
    v16h a1 = load_frag_a(qf, 40, 16, 0, lane);
#pragma unroll
    for (int j = 0; j < 8; ++j) {
      v16h bf = load_frag_b(kT, 40, 0, j * 16, lane);
      v8f d0 = {}; d0 = WMMA_F16(a0, bf, d0);
      v8f d1 = {}; d1 = WMMA_F16(a1, bf, d1);
      store_frag_lds(lg, 128, 0, j * 16, d0, lane);
      store_frag_lds(lg, 128, 16, j * 16, d1, lane);
    }
  }
  __syncthreads();
  // softmax (+pair bias, +key mask): one row per lane, float4 LDS traffic
  {
    const int row = lane;
    const float sc = 0.17677669529663687f;  // HD^-0.5
    const float* plr = pl + ((long)(s * QQ + row) * KK) * (NBB * NHH) + bIdx * NHH + h;
    const unsigned char* km = keys_mask + (long)s * KK;
    float mx = -1e30f;
    for (int kk = 0; kk < KK; kk += 4) {
      float4 l4 = *(const float4*)&lg[row * 128 + kk];
      l4.x = km[kk + 0] ? l4.x * sc + plr[(long)(kk + 0) * (NBB * NHH)] : -1e9f;
      l4.y = km[kk + 1] ? l4.y * sc + plr[(long)(kk + 1) * (NBB * NHH)] : -1e9f;
      l4.z = km[kk + 2] ? l4.z * sc + plr[(long)(kk + 2) * (NBB * NHH)] : -1e9f;
      l4.w = km[kk + 3] ? l4.w * sc + plr[(long)(kk + 3) * (NBB * NHH)] : -1e9f;
      *(float4*)&lg[row * 128 + kk] = l4;
      mx = fmaxf(mx, fmaxf(fmaxf(l4.x, l4.y), fmaxf(l4.z, l4.w)));
    }
    float sum = 0.f;
    for (int kk = 0; kk < KK; kk += 4) {
      float4 l4 = *(const float4*)&lg[row * 128 + kk];
      l4.x = __expf(l4.x - mx); l4.y = __expf(l4.y - mx);
      l4.z = __expf(l4.z - mx); l4.w = __expf(l4.w - mx);
      *(float4*)&lg[row * 128 + kk] = l4;
      sum += l4.x + l4.y + l4.z + l4.w;
    }
    float inv = 1.f / sum;
    for (int kk = 0; kk < KK; kk += 4) {
      float4 l4 = *(const float4*)&lg[row * 128 + kk];
      *(v4h*)&pf[row * 136 + kk] = cvt4(l4.x * inv, l4.y * inv, l4.z * inv, l4.w * inv);
    }
  }
  __syncthreads();
  // O = P(32x128) @ V(128x32): 4 K-steps, 2x2 tiles; B read n-major from vT (ld=136).
  v8f acc[2][2] = {};
#pragma unroll
  for (int kst = 0; kst < 4; ++kst) {
    v16h p0 = load_frag_a(pf, 136, 0, kst * 32, lane);
    v16h p1 = load_frag_a(pf, 136, 16, kst * 32, lane);
    v16h v0 = load_frag_b(vT, 136, kst * 32, 0, lane);
    v16h v1 = load_frag_b(vT, 136, kst * 32, 16, lane);
    acc[0][0] = WMMA_F16(p0, v0, acc[0][0]);
    acc[0][1] = WMMA_F16(p0, v1, acc[0][1]);
    acc[1][0] = WMMA_F16(p1, v0, acc[1][0]);
    acc[1][1] = WMMA_F16(p1, v1, acc[1][1]);
  }
  float* os = og + (long)s * QQ * CC + h * HDD;
  const int n = lane & 15, mb = (lane >> 4) << 3;
#pragma unroll
  for (int i = 0; i < 2; ++i)
#pragma unroll
    for (int j = 0; j < 2; ++j)
#pragma unroll
      for (int r = 0; r < 8; ++r)
        os[(i * 16 + mb + r) * CC + j * 16 + n] = acc[i][j][r];
}

// ---------------- small VALU kernels ----------------------------------------
__device__ __forceinline__ float block_reduce128(float v, float* red) {
  int tid = threadIdx.x;
  red[tid] = v;
  __syncthreads();
  for (int s = 64; s > 0; s >>= 1) {
    if (tid < s) red[tid] += red[tid + s];
    __syncthreads();
  }
  float r = red[0];
  __syncthreads();
  return r;
}

// y = ln(x_row) * scale[c]   (one 128-wide row per block)
__global__ __launch_bounds__(128) void ln_rows_kernel(const float* __restrict__ x,
                                                      const float* __restrict__ scale,
                                                      float* __restrict__ y) {
  __shared__ float red[128];
  long row = blockIdx.x; int c = threadIdx.x;
  float v = x[row * CC + c];
  float mu = block_reduce128(v, red) * (1.f / CC);
  float d = v - mu;
  float var = block_reduce128(d * d, red) * (1.f / CC);
  y[row * CC + c] = d * rsqrtf(var + 1e-5f) * scale[c];
}

// y = sigmoid(t1 + bscale[c]) * ln(x_row) + t2
__global__ __launch_bounds__(128) void modulate_kernel(const float* __restrict__ t1,
                                                       const float* __restrict__ bscale,
                                                       const float* __restrict__ x,
                                                       const float* __restrict__ t2,
                                                       float* __restrict__ y) {
  __shared__ float red[128];
  long row = blockIdx.x; int c = threadIdx.x;
  float xv = x[row * CC + c];
  float mu = block_reduce128(xv, red) * (1.f / CC);
  float d = xv - mu;
  float var = block_reduce128(d * d, red) * (1.f / CC);
  float ln = d * rsqrtf(var + 1e-5f);
  float s = 1.f / (1.f + __expf(-(t1[row * CC + c] + bscale[c])));
  y[row * CC + c] = s * ln + t2[row * CC + c];
}

// x[s,q,:] = (gather(tok, a2q) + skip) * queries_mask
__global__ __launch_bounds__(128) void build_x_kernel(const float* __restrict__ tok,
                                                      const float* __restrict__ skip,
                                                      const int* __restrict__ idx,
                                                      const unsigned char* __restrict__ msk,
                                                      const unsigned char* __restrict__ qmsk,
                                                      float* __restrict__ x) {
  long row = blockIdx.x; int c = threadIdx.x;
  float t = msk[row] ? tok[(long)(idx[row] / AA) * CC + c] : 0.f;
  x[row * CC + c] = (t + skip[row * CC + c]) * (qmsk[row] ? 1.f : 0.f);
}

// keys_act[s,k,:] = gather(x, q2k) * mask
__global__ __launch_bounds__(128) void gather_keys_kernel(const float* __restrict__ x,
                                                          const int* __restrict__ idx,
                                                          const unsigned char* __restrict__ msk,
                                                          float* __restrict__ y) {
  long row = blockIdx.x; int c = threadIdx.x;
  y[row * CC + c] = msk[row] ? x[(long)idx[row] * CC + c] : 0.f;
}

// pair logits: pl[r, 0:12] = ln(pair_cond[r,:], pair_ln_scale) @ w_pair
__global__ __launch_bounds__(128) void pair_kernel(const float* __restrict__ pc,
                                                   const float* __restrict__ plns,
                                                   const float* __restrict__ wpair,
                                                   float* __restrict__ pl) {
  long r = (long)blockIdx.x * 128 + threadIdx.x;
  if (r >= (long)SS * QQ * KK) return;
  const float* p = pc + r * CPP;
  float mu = 0.f;
#pragma unroll
  for (int i = 0; i < CPP; ++i) mu += p[i];
  mu *= (1.f / CPP);
  float var = 0.f;
#pragma unroll
  for (int i = 0; i < CPP; ++i) { float d = p[i] - mu; var += d * d; }
  var *= (1.f / CPP);
  float inv = rsqrtf(var + 1e-5f);
  float ln[CPP];
#pragma unroll
  for (int i = 0; i < CPP; ++i) ln[i] = (p[i] - mu) * inv * plns[i];
#pragma unroll
  for (int j = 0; j < NBB * NHH; ++j) {
    float a = 0.f;
#pragma unroll
    for (int i = 0; i < CPP; ++i) a += ln[i] * wpair[i * (NBB * NHH) + j];
    pl[r * (NBB * NHH) + j] = a;
  }
}

// x += sigmoid(g + bias[c]) * upd
__global__ __launch_bounds__(256) void apply_gate_kernel(float* __restrict__ x,
                                                         const float* __restrict__ g,
                                                         const float* __restrict__ bias,
                                                         const float* __restrict__ upd, long n) {
  long i = (long)blockIdx.x * 256 + threadIdx.x;
  if (i >= n) return;
  int c = (int)(i & (CC - 1));
  float s = 1.f / (1.f + __expf(-(g[i] + bias[c])));
  x[i] += s * upd[i];
}

// sw[row, j] = swish(h[row, j]) * h[row, FH + j]
__global__ __launch_bounds__(256) void swiglu_kernel(const float* __restrict__ h,
                                                     float* __restrict__ sw) {
  long row = blockIdx.x; int j = threadIdx.x;
  float a = h[row * (2 * FHH) + j];
  float g = h[row * (2 * FHH) + FHH + j];
  sw[row * FHH + j] = (a / (1.f + __expf(-a))) * g;
}

// y = ln(x_row * qm) * final_scale
__global__ __launch_bounds__(128) void final_ln_kernel(const float* __restrict__ x,
                                                       const unsigned char* __restrict__ qmsk,
                                                       const float* __restrict__ scale,
                                                       float* __restrict__ y) {
  __shared__ float red[128];
  long row = blockIdx.x; int c = threadIdx.x;
  float v = x[row * CC + c] * (qmsk[row] ? 1.f : 0.f);
  float mu = block_reduce128(v, red) * (1.f / CC);
  float d = v - mu;
  float var = block_reduce128(d * d, red) * (1.f / CC);
  y[row * CC + c] = d * rsqrtf(var + 1e-5f) * scale[c];
}

// pos[row, 0:3] = xln[row,:] @ w_pos
__global__ __launch_bounds__(256) void pos_kernel(const float* __restrict__ xln,
                                                  const float* __restrict__ wpos,
                                                  float* __restrict__ pos) {
  long r = (long)blockIdx.x * 256 + threadIdx.x;
  if (r >= (long)SS * QQ) return;
  const float* xr = xln + r * CC;
  float a0 = 0.f, a1 = 0.f, a2 = 0.f;
  for (int c = 0; c < CC; ++c) {
    float v = xr[c];
    a0 += v * wpos[c * 3 + 0];
    a1 += v * wpos[c * 3 + 1];
    a2 += v * wpos[c * 3 + 2];
  }
  pos[r * 3 + 0] = a0; pos[r * 3 + 1] = a1; pos[r * 3 + 2] = a2;
}

// out[t,a,0:3] = gather(pos, q2a) * mask
__global__ __launch_bounds__(256) void scatter_kernel(const float* __restrict__ pos,
                                                      const int* __restrict__ idx,
                                                      const unsigned char* __restrict__ msk,
                                                      float* __restrict__ out) {
  long r = (long)blockIdx.x * 256 + threadIdx.x;
  if (r >= (long)TT * AA) return;
  float m = msk[r] ? 1.f : 0.f;
  long id = idx[r];
  out[r * 3 + 0] = pos[id * 3 + 0] * m;
  out[r * 3 + 1] = pos[id * 3 + 1] * m;
  out[r * 3 + 2] = pos[id * 3 + 2] * m;
}

// ---------------- host orchestration ----------------------------------------
extern "C" void kernel_launch(void* const* d_in, const int* in_sizes, int n_in,
                              void* d_out, int out_size, void* d_ws, size_t ws_size,
                              hipStream_t stream) {
  (void)in_sizes; (void)n_in; (void)out_size; (void)ws_size;
  const float* token_act      = (const float*)d_in[0];
  const float* skip           = (const float*)d_in[1];
  const float* qsc            = (const float*)d_in[2];
  const float* ksc            = (const float*)d_in[3];
  const float* pair_cond      = (const float*)d_in[4];
  const float* w_proj         = (const float*)d_in[5];
  const float* pair_ln_scale  = (const float*)d_in[6];
  const float* w_pair         = (const float*)d_in[7];
  const float* qln_cond_scale = (const float*)d_in[8];
  const float* qln_wscale     = (const float*)d_in[9];
  const float* qln_bscale     = (const float*)d_in[10];
  const float* qln_wbias      = (const float*)d_in[11];
  const float* kln_cond_scale = (const float*)d_in[12];
  const float* kln_wscale     = (const float*)d_in[13];
  const float* kln_bscale     = (const float*)d_in[14];
  const float* kln_wbias      = (const float*)d_in[15];
  const float* wq             = (const float*)d_in[16];
  const float* wk             = (const float*)d_in[17];
  const float* wv             = (const float*)d_in[18];
  const float* wgate          = (const float*)d_in[19];
  const float* bgate          = (const float*)d_in[20];
  const float* wout           = (const float*)d_in[21];
  const float* tln_cond_scale = (const float*)d_in[22];
  const float* tln_wscale     = (const float*)d_in[23];
  const float* tln_bscale     = (const float*)d_in[24];
  const float* tln_wbias      = (const float*)d_in[25];
  const float* wtrans_in      = (const float*)d_in[26];
  const float* wtrans_out     = (const float*)d_in[27];
  const float* wtgate         = (const float*)d_in[28];
  const float* btgate         = (const float*)d_in[29];
  const float* final_ln_scale = (const float*)d_in[30];
  const float* w_pos          = (const float*)d_in[31];
  const int*   a2q_idx        = (const int*)d_in[32];
  const unsigned char* a2q_mask = (const unsigned char*)d_in[33];
  const int*   q2k_idx        = (const int*)d_in[34];
  const unsigned char* q2k_mask = (const unsigned char*)d_in[35];
  const int*   q2a_idx        = (const int*)d_in[36];
  const unsigned char* q2a_mask = (const unsigned char*)d_in[37];
  const unsigned char* queries_mask = (const unsigned char*)d_in[38];
  const unsigned char* keys_mask    = (const unsigned char*)d_in[39];

  // workspace carve-up (floats)
  float* w = (float*)d_ws;
  size_t o = 0;
  auto alloc = [&](size_t n) { float* p = w + o; o += n; return p; };
  const long RQ = (long)SS * QQ;      // 16384
  const long RK = (long)SS * KK;      // 65536
  float* tok  = alloc((size_t)TT * CC);
  float* x    = alloc(RQ * CC);
  float* pl   = alloc(RQ * KK * NBB * NHH);
  float* cq   = alloc(RQ * CC);
  float* ck   = alloc(RK * CC);
  float* keys = alloc(RK * CC);
  float* t1   = alloc(RQ * CC);
  float* t2   = alloc(RQ * CC);
  float* kt1  = alloc(RK * CC);
  float* kt2  = alloc(RK * CC);
  float* qn   = alloc(RQ * CC);
  float* kn   = alloc(RK * CC);
  float* qb   = alloc(RQ * CC);
  float* kb   = alloc(RK * CC);
  float* vb   = alloc(RK * CC);
  float* ob   = alloc(RQ * CC);
  float* g1   = alloc(RQ * CC);
  float* xo   = alloc(RQ * CC);
  float* ct   = alloc(RQ * CC);
  float* xt   = alloc(RQ * CC);
  float* hbuf = alloc(RQ * 2 * FHH);
  float* sw   = alloc(RQ * FHH);
  float* outt = alloc(RQ * CC);
  float* tg   = alloc(RQ * CC);
  float* xln  = alloc(RQ * CC);
  float* pos  = alloc(RQ * 3);

  auto gemm = [&](const float* A, const float* B, float* D, int M, int N, int Kd) {
    dim3 g(N / 64, M / 64);
    wmma_gemm_kernel<<<g, 128, 0, stream>>>(A, B, D, M, N, Kd);
  };

  // token projection + gathered query activations + pair logits
  gemm(token_act, w_proj, tok, TT, CC, CTT);
  build_x_kernel<<<(int)RQ, 128, 0, stream>>>(tok, skip, a2q_idx, a2q_mask, queries_mask, x);
  pair_kernel<<<(int)((RQ * KK + 127) / 128), 128, 0, stream>>>(pair_cond, pair_ln_scale, w_pair, pl);

  for (int b = 0; b < NBB; ++b) {
    gather_keys_kernel<<<(int)RK, 128, 0, stream>>>(x, q2k_idx, q2k_mask, keys);
    ln_rows_kernel<<<(int)RQ, 128, 0, stream>>>(qsc, qln_cond_scale + (long)b * CC, cq);
    ln_rows_kernel<<<(int)RK, 128, 0, stream>>>(ksc, kln_cond_scale + (long)b * CC, ck);
    // adaptive LN (query side)
    gemm(cq, qln_wscale + (long)b * CC * CC, t1, (int)RQ, CC, CC);
    gemm(cq, qln_wbias + (long)b * CC * CC, t2, (int)RQ, CC, CC);
    modulate_kernel<<<(int)RQ, 128, 0, stream>>>(t1, qln_bscale + (long)b * CC, x, t2, qn);
    // adaptive LN (key side)
    gemm(ck, kln_wscale + (long)b * CC * CC, kt1, (int)RK, CC, CC);
    gemm(ck, kln_wbias + (long)b * CC * CC, kt2, (int)RK, CC, CC);
    modulate_kernel<<<(int)RK, 128, 0, stream>>>(kt1, kln_bscale + (long)b * CC, keys, kt2, kn);
    // Q/K/V projections
    gemm(qn, wq + (long)b * CC * (NHH * HDD), qb, (int)RQ, NHH * HDD, CC);
    gemm(kn, wk + (long)b * CC * (NHH * HDD), kb, (int)RK, NHH * HDD, CC);
    gemm(kn, wv + (long)b * CC * (NHH * HDD), vb, (int)RK, NHH * HDD, CC);
    // fused attention per (subset, head)
    attn_kernel<<<SS * NHH, 32, 0, stream>>>(qb, kb, vb, pl, keys_mask, ob, b);
    // gated residual
    gemm(cq, wgate + (long)b * CC * CC, g1, (int)RQ, CC, CC);
    gemm(ob, wout + (long)b * (NHH * HDD) * CC, xo, (int)RQ, CC, NHH * HDD);
    apply_gate_kernel<<<(int)((RQ * CC + 255) / 256), 256, 0, stream>>>(x, g1, bgate + (long)b * CC, xo, RQ * CC);
    // transition: adaptive LN -> swiglu -> gated residual
    ln_rows_kernel<<<(int)RQ, 128, 0, stream>>>(qsc, tln_cond_scale + (long)b * CC, ct);
    gemm(ct, tln_wscale + (long)b * CC * CC, t1, (int)RQ, CC, CC);
    gemm(ct, tln_wbias + (long)b * CC * CC, t2, (int)RQ, CC, CC);
    modulate_kernel<<<(int)RQ, 128, 0, stream>>>(t1, tln_bscale + (long)b * CC, x, t2, xt);
    gemm(xt, wtrans_in + (long)b * CC * (2 * FHH), hbuf, (int)RQ, 2 * FHH, CC);
    swiglu_kernel<<<(int)RQ, FHH, 0, stream>>>(hbuf, sw);
    gemm(sw, wtrans_out + (long)b * FHH * CC, outt, (int)RQ, CC, FHH);
    gemm(ct, wtgate + (long)b * CC * CC, tg, (int)RQ, CC, CC);
    apply_gate_kernel<<<(int)((RQ * CC + 255) / 256), 256, 0, stream>>>(x, tg, btgate + (long)b * CC, outt, RQ * CC);
  }

  final_ln_kernel<<<(int)RQ, 128, 0, stream>>>(x, queries_mask, final_ln_scale, xln);
  pos_kernel<<<(int)((RQ + 255) / 256), 256, 0, stream>>>(xln, w_pos, pos);
  scatter_kernel<<<(int)(((long)TT * AA + 255) / 256), 256, 0, stream>>>(pos, q2a_idx, q2a_mask, (float*)d_out);
}
